// RPEMultiHeadAttention_89189290869050
// MI455X (gfx1250) — matrix operand order repaired and checked by reading, not compiled, not run on hardware
//
#include <hip/hip_runtime.h>
#include <math.h>

typedef __attribute__((ext_vector_type(16))) __bf16 v16bf;
typedef __attribute__((ext_vector_type(8)))  __bf16 v8bf;
typedef __attribute__((ext_vector_type(4)))  __bf16 v4bf;
typedef __attribute__((ext_vector_type(8)))  float  v8f;
typedef __attribute__((ext_vector_type(4)))  float  v4f;

#define D_MODEL 1024
#define N_HEADS 16
#define D_HEAD  64
#define SEQ_T   1024
#define BATCH_N 4
#define ATT_SCALE 0.125f      /* 64^-0.5 */
#define R_ROWS  2048          /* 2T-1 = 2047 padded */

static __device__ __forceinline__ v8f wmma_bf16(v16bf a, v16bf b, v8f c) {
  // D = A(16x32 bf16) * B(32x16 bf16) + C(f32)
  return __builtin_amdgcn_wmma_f32_16x16x32_bf16(false, a, false, b, (short)0, c,
                                                 false, false);
}
static __device__ __forceinline__ v8bf  load8 (const __bf16* p) { return *(const v8bf*)p;  }
static __device__ __forceinline__ v16bf load16(const __bf16* p) { return *(const v16bf*)p; }
static __device__ __forceinline__ v16bf combine8(v8bf lo, v8bf hi) {
  v16bf r;
#pragma unroll
  for (int i = 0; i < 8; ++i) { r[i] = lo[i]; r[i + 8] = hi[i]; }
  return r;
}

// -------------------- fp32 -> bf16 conversion (4 elems/thread) --------------------
__global__ __launch_bounds__(256)
void f32_to_bf16_kernel(const float* __restrict__ src, __bf16* __restrict__ dst, int count4) {
  const int i = blockIdx.x * blockDim.x + threadIdx.x;
  if (i < count4) {
    const v4f s = ((const v4f*)src)[i];
    v4bf d;
#pragma unroll
    for (int j = 0; j < 4; ++j) d[j] = (__bf16)s[j];
    ((v4bf*)dst)[i] = d;
  }
}

// -------------------- generic WMMA GEMM: C = A * B^T + bias1 + bias2 --------------------
// A: (M,K) bf16 row-major. Rows >= Mreal are CLAMPED to row Mreal-1 (only the padded
//    R-projection row 2047 hits this; its product feeds a never-selected rel-shift slot).
// B: (Nn,K) bf16 row-major.
// Each wave computes a 16(M) x 32(N) strip (2 accumulators sharing one A tile);
// 8 waves cover 64x64. k-loop is software-pipelined (double-buffered operands).
__global__ __launch_bounds__(256)
void wmma_gemm_bt_kernel(const __bf16* __restrict__ A, int lda, int Mreal,
                         const __bf16* __restrict__ B, int ldb,
                         const float* __restrict__ bias1, const float* __restrict__ bias2,
                         __bf16* __restrict__ Cb, float* __restrict__ Cf,
                         int M, int Nn, int Kdim, int Tdim, int storeT) {
  const int tid  = threadIdx.x;
  const int wave = tid >> 5;
  const int lane = tid & 31;
  const int lm   = lane & 15;
  const int hi   = lane >> 4;
  const int tileM  = blockIdx.y * 4 + (wave >> 1);    // 16-row tile
  const int tileN2 = blockIdx.x * 2 + (wave & 1);     // 32-col strip
  if (tileM * 16 >= M || tileN2 * 32 >= Nn) return;

  const int m    = tileM * 16 + lm;
  const int mrow = (m < Mreal) ? m : (Mreal - 1);     // clamp, no divergence
  const size_t abase  = (size_t)mrow * lda + (hi ? 16 : 0);
  const size_t b0base = (size_t)(tileN2 * 32 +      lm) * ldb + (hi ? 16 : 0);
  const size_t b1base = (size_t)(tileN2 * 32 + 16 + lm) * ldb + (hi ? 16 : 0);

  v8f acc0 = {};
  v8f acc1 = {};
  // prologue loads (k0 = 0)
  v16bf a  = combine8(load8(A + abase), load8(A + abase + 8));
  v16bf b0 = load16(B + b0base);
  v16bf b1 = load16(B + b1base);
  int k0 = 0;
  for (; k0 + 32 < Kdim; k0 += 32) {
    const int kn = k0 + 32;
    v16bf an  = combine8(load8(A + abase + kn), load8(A + abase + kn + 8));
    v16bf b0n = load16(B + b0base + kn);
    v16bf b1n = load16(B + b1base + kn);
    acc0 = wmma_bf16(a, b0, acc0);
    acc1 = wmma_bf16(a, b1, acc1);
    a = an; b0 = b0n; b1 = b1n;
  }
  acc0 = wmma_bf16(a, b0, acc0);
  acc1 = wmma_bf16(a, b1, acc1);

#pragma unroll
  for (int g = 0; g < 2; ++g) {
    const int   colg = tileN2 * 32 + g * 16 + lm;
    const float b1s = bias1 ? bias1[colg] : 0.0f;
    const float b2s = bias2 ? bias2[colg] : 0.0f;
    const v8f&  acc = g ? acc1 : acc0;
#pragma unroll
    for (int r = 0; r < 8; ++r) {
      const int rowg = tileM * 16 + r + 8 * hi;   // C layout: VGPR r -> row r / r+8
      const float v = acc[r] + b1s + b2s;
      if (Cf) {
        Cf[(size_t)rowg * Nn + colg] = v;
      } else if (storeT) {
        const int nn = rowg / Tdim, tt = rowg % Tdim;
        Cb[((size_t)nn * Nn + colg) * Tdim + tt] = (__bf16)v;   // Vt[n][d][t]
      } else {
        Cb[(size_t)rowg * Nn + colg] = (__bf16)v;
      }
    }
  }
}

// -------------------- flash-style relative-position attention --------------------
// grid (T/64, H, N); 256 threads = 8 waves; 64 query rows x 64-key panels.
__global__ __launch_bounds__(256)
void rpe_attention_kernel(const __bf16* __restrict__ Qu, const __bf16* __restrict__ Qv,
                          const __bf16* __restrict__ Kb, const __bf16* __restrict__ Vt,
                          const __bf16* __restrict__ Rb, __bf16* __restrict__ Ob) {
  __shared__ __align__(32) float  S[64 * 64];     // 16 KB scaled scores
  __shared__ __align__(32) __bf16 P[64 * 64];     //  8 KB probs (bf16)
  __shared__ float partmax[64 * 4], partsum[64 * 4];
  __shared__ float m_run[64], l_run[64], alpha_s[64];

  const int tid  = threadIdx.x;
  const int wave = tid >> 5;
  const int lane = tid & 31;
  const int lm   = lane & 15;
  const int hi   = lane >> 4;
  const int qbase = blockIdx.x * 64;
  const int h     = blockIdx.y;
  const int n     = blockIdx.z;

  // this wave's score tiles: fixed ct = wave&3, mt in {wave>>2, wave>>2 + 2}
  const int ct  = wave & 3;
  const int mt0 = wave >> 2;

  if (tid < 64) { m_run[tid] = -1e30f; l_run[tid] = 0.0f; }
  __syncthreads();

  // hoist panel-invariant Q(+u)/Q(+v) A-operands into registers (8 x v16bf)
  v16bf qU[2][2], qV[2][2];
#pragma unroll
  for (int ti = 0; ti < 2; ++ti) {
    const int rowbase = qbase + (mt0 + 2 * ti) * 16;
#pragma unroll
    for (int kk = 0; kk < 2; ++kk) {
      const size_t aoff = (size_t)(n * SEQ_T + rowbase + lm) * D_MODEL
                        + h * D_HEAD + kk * 32 + (hi ? 16 : 0);
      qU[ti][kk] = combine8(load8(Qu + aoff), load8(Qu + aoff + 8));
      qV[ti][kk] = combine8(load8(Qv + aoff), load8(Qv + aoff + 8));
    }
  }

  v8f acc0 = {};
  v8f acc1 = {};

  for (int p = 0; p < SEQ_T / 64; ++p) {
    const int pb = p * 64;
    if (p + 1 < SEQ_T / 64) {   // warm next panel (global_prefetch_b8)
      __builtin_prefetch(Kb + ((size_t)(n * SEQ_T + pb + 64 + lane) * D_MODEL + h * D_HEAD), 0, 1);
      __builtin_prefetch(Vt + ((size_t)(n * D_MODEL + h * D_HEAD + lane) * SEQ_T + pb + 64), 0, 1);
    }
    // ---- scores: AC + rel-shifted BD ----
#pragma unroll
    for (int ti = 0; ti < 2; ++ti) {
      const int mt = mt0 + 2 * ti;
      const int rowbase = qbase + mt * 16;    // absolute query i base
      const int colbase = pb + ct * 16;       // absolute key   j base
      const int rbase   = rowbase - colbase + 1008; // raw band: r = i-j+1023, c_rel = di-dj+15
      v8f ac = {}; v8f r0 = {}; v8f r1 = {};
#pragma unroll
      for (int kk = 0; kk < 2; ++kk) {
        const int k0 = kk * 32;
        const size_t koff = (size_t)(n * SEQ_T + colbase + lm) * D_MODEL + h * D_HEAD + k0 + (hi ? 16 : 0);
        ac = wmma_bf16(qU[ti][kk], load16(Kb + koff), ac);
        const size_t roff0 = (size_t)(rbase +      lm) * D_MODEL + h * D_HEAD + k0 + (hi ? 16 : 0);
        const size_t roff1 = (size_t)(rbase + 16 + lm) * D_MODEL + h * D_HEAD + k0 + (hi ? 16 : 0);
        r0 = wmma_bf16(qV[ti][kk], load16(Rb + roff0), r0);
        r1 = wmma_bf16(qV[ti][kk], load16(Rb + roff1), r1);
      }
      // rel-shift select: BD[di,dj] = raw[di, di-dj+15]; same VGPR index, pure lane permute
#pragma unroll
      for (int r = 0; r < 8; ++r) {
        const int di  = r + 8 * hi;
        const int c   = di - lm + 15;                 // 0..30
        const int src = (c & 15) + (hi << 4);
        const float v0 = __shfl(r0[r], src, 32);
        const float v1 = __shfl(r1[r], src, 32);
        const float bd = (c < 16) ? v0 : v1;
        S[(mt * 16 + di) * 64 + ct * 16 + lm] = (ac[r] + bd) * ATT_SCALE;
      }
    }
    __syncthreads();
    { // partial row max (4 threads/row)
      const int row = tid >> 2, part = tid & 3;
      float mx = -1e30f;
      const float* sp = &S[row * 64 + part * 16];
#pragma unroll
      for (int i = 0; i < 16; ++i) mx = fmaxf(mx, sp[i]);
      partmax[row * 4 + part] = mx;
    }
    __syncthreads();
    if (tid < 64) { // online-softmax stats
      const float pm = fmaxf(fmaxf(partmax[tid * 4], partmax[tid * 4 + 1]),
                             fmaxf(partmax[tid * 4 + 2], partmax[tid * 4 + 3]));
      const float mo = m_run[tid];
      const float mn = fmaxf(mo, pm);
      alpha_s[tid] = __expf(mo - mn);
      m_run[tid]   = mn;
    }
    __syncthreads();
    { // exp, pack bf16 probs, partial sums
      const int row = tid >> 2, part = tid & 3;
      const float mn = m_run[row];
      const int base = row * 64 + part * 16;
      float s = 0.0f;
#pragma unroll
      for (int i = 0; i < 16; ++i) {
        const float pe = __expf(S[base + i] - mn);
        P[base + i] = (__bf16)pe;
        s += pe;
      }
      partsum[row * 4 + part] = s;
    }
    __syncthreads();
    if (tid < 64) {
      l_run[tid] = l_run[tid] * alpha_s[tid]
                 + partsum[tid * 4] + partsum[tid * 4 + 1]
                 + partsum[tid * 4 + 2] + partsum[tid * 4 + 3];
    }
    __syncthreads();
    // ---- rescale accumulators, O += P * V (2 output tiles per wave) ----
#pragma unroll
    for (int oi = 0; oi < 2; ++oi) {
      const int ot = wave * 2 + oi;
      const int mo = ot >> 2, no = ot & 3;
      v8f a = (oi == 0) ? acc0 : acc1;
#pragma unroll
      for (int r = 0; r < 8; ++r) a[r] *= alpha_s[mo * 16 + r + 8 * hi];
#pragma unroll
      for (int k0 = 0; k0 < 64; k0 += 32) {
        const __bf16* ap = &P[(mo * 16 + lm) * 64 + k0 + (hi ? 16 : 0)];
        v16bf av = combine8(*(const v8bf*)ap, *(const v8bf*)(ap + 8));
        const size_t voff = (size_t)(n * D_MODEL + h * D_HEAD + no * 16 + lm) * SEQ_T
                          + pb + k0 + (hi ? 16 : 0);
        a = wmma_bf16(av, load16(Vt + voff), a);
      }
      if (oi == 0) acc0 = a; else acc1 = a;
    }
    __syncthreads();
  }
  // ---- normalize and store per-head output (bf16) ----
#pragma unroll
  for (int oi = 0; oi < 2; ++oi) {
    const int ot = wave * 2 + oi;
    const int mo = ot >> 2, no = ot & 3;
    v8f a = (oi == 0) ? acc0 : acc1;
#pragma unroll
    for (int r = 0; r < 8; ++r) {
      const int rl = mo * 16 + r + 8 * hi;
      const float inv = 1.0f / l_run[rl];
      const size_t off = (size_t)(n * SEQ_T + qbase + rl) * D_MODEL + h * D_HEAD + no * 16 + lm;
      Ob[off] = (__bf16)(a[r] * inv);
    }
  }
}

// -------------------- host launcher --------------------
extern "C" void kernel_launch(void* const* d_in, const int* in_sizes, int n_in,
                              void* d_out, int out_size, void* d_ws, size_t ws_size,
                              hipStream_t stream) {
  const float* x  = (const float*)d_in[0];
  const float* pe = (const float*)d_in[1];
  const float* Wq = (const float*)d_in[2];
  const float* bq = (const float*)d_in[3];
  const float* Wk = (const float*)d_in[4];
  const float* bk = (const float*)d_in[5];
  const float* Wv = (const float*)d_in[6];
  const float* bv = (const float*)d_in[7];
  const float* Wo = (const float*)d_in[8];
  const float* bo = (const float*)d_in[9];
  const float* Wr = (const float*)d_in[10];
  const float* u  = (const float*)d_in[11];   // (16,64) flat == per-column bias
  const float* v  = (const float*)d_in[12];
  float* out = (float*)d_out;
  (void)in_sizes; (void)n_in; (void)out_size; (void)ws_size;

  const size_t NT = (size_t)BATCH_N * SEQ_T;  // 4096
  char* w = (char*)d_ws;
  __bf16* xb  = (__bf16*)w; w += NT * D_MODEL * 2;
  __bf16* peb = (__bf16*)w; w += (size_t)R_ROWS * D_MODEL * 2;
  __bf16* wqb = (__bf16*)w; w += (size_t)D_MODEL * D_MODEL * 2;
  __bf16* wkb = (__bf16*)w; w += (size_t)D_MODEL * D_MODEL * 2;
  __bf16* wvb = (__bf16*)w; w += (size_t)D_MODEL * D_MODEL * 2;
  __bf16* wob = (__bf16*)w; w += (size_t)D_MODEL * D_MODEL * 2;
  __bf16* wrb = (__bf16*)w; w += (size_t)D_MODEL * D_MODEL * 2;
  __bf16* Qu  = (__bf16*)w; w += NT * D_MODEL * 2;
  __bf16* Qv  = (__bf16*)w; w += NT * D_MODEL * 2;
  __bf16* Kb  = (__bf16*)w; w += NT * D_MODEL * 2;
  __bf16* Vt  = (__bf16*)w; w += NT * D_MODEL * 2;   // Vt[n][d][t]
  __bf16* Rb  = (__bf16*)w; w += (size_t)R_ROWS * D_MODEL * 2;
  __bf16* Ao  = (__bf16*)w; w += NT * D_MODEL * 2;

  const dim3 blk(256);
  auto cdiv4 = [](size_t elems) { return (unsigned)((elems / 4 + 255) / 256); };

  // fp32 -> bf16 (vectorized x4)
  f32_to_bf16_kernel<<<cdiv4(NT * D_MODEL), blk, 0, stream>>>(x, xb, (int)(NT * D_MODEL / 4));
  f32_to_bf16_kernel<<<cdiv4((size_t)2047 * D_MODEL), blk, 0, stream>>>(pe, peb, 2047 * D_MODEL / 4);
  f32_to_bf16_kernel<<<cdiv4((size_t)D_MODEL * D_MODEL), blk, 0, stream>>>(Wq, wqb, D_MODEL * D_MODEL / 4);
  f32_to_bf16_kernel<<<cdiv4((size_t)D_MODEL * D_MODEL), blk, 0, stream>>>(Wk, wkb, D_MODEL * D_MODEL / 4);
  f32_to_bf16_kernel<<<cdiv4((size_t)D_MODEL * D_MODEL), blk, 0, stream>>>(Wv, wvb, D_MODEL * D_MODEL / 4);
  f32_to_bf16_kernel<<<cdiv4((size_t)D_MODEL * D_MODEL), blk, 0, stream>>>(Wo, wob, D_MODEL * D_MODEL / 4);
  f32_to_bf16_kernel<<<cdiv4((size_t)D_MODEL * D_MODEL), blk, 0, stream>>>(Wr, wrb, D_MODEL * D_MODEL / 4);

  // projections (WMMA GEMMs): block covers 64x64
  const dim3 g_proj(D_MODEL / 64, (unsigned)(NT / 64));  // (16,64)
  wmma_gemm_bt_kernel<<<g_proj, blk, 0, stream>>>(xb, D_MODEL, (int)NT, wqb, D_MODEL,
      bq, u, Qu, nullptr, (int)NT, D_MODEL, D_MODEL, SEQ_T, 0);
  wmma_gemm_bt_kernel<<<g_proj, blk, 0, stream>>>(xb, D_MODEL, (int)NT, wqb, D_MODEL,
      bq, v, Qv, nullptr, (int)NT, D_MODEL, D_MODEL, SEQ_T, 0);
  wmma_gemm_bt_kernel<<<g_proj, blk, 0, stream>>>(xb, D_MODEL, (int)NT, wkb, D_MODEL,
      bk, nullptr, Kb, nullptr, (int)NT, D_MODEL, D_MODEL, SEQ_T, 0);
  wmma_gemm_bt_kernel<<<g_proj, blk, 0, stream>>>(xb, D_MODEL, (int)NT, wvb, D_MODEL,
      bv, nullptr, Vt, nullptr, (int)NT, D_MODEL, D_MODEL, SEQ_T, 1);   // transposed store
  const dim3 g_r(D_MODEL / 64, R_ROWS / 64);             // (16,32)
  wmma_gemm_bt_kernel<<<g_r, blk, 0, stream>>>(peb, D_MODEL, 2047, wrb, D_MODEL,
      nullptr, nullptr, Rb, nullptr, R_ROWS, D_MODEL, D_MODEL, SEQ_T, 0);

  // flash attention with relative-position shift
  const dim3 g_att(SEQ_T / 64, N_HEADS, BATCH_N);
  rpe_attention_kernel<<<g_att, blk, 0, stream>>>(Qu, Qv, Kb, Vt, Rb, Ao);

  // output projection -> fp32 d_out
  wmma_gemm_bt_kernel<<<g_proj, blk, 0, stream>>>(Ao, D_MODEL, (int)NT, wob, D_MODEL,
      bo, nullptr, nullptr, out, (int)NT, D_MODEL, D_MODEL, SEQ_T, 0);
}
